// SparseImVoxelNeck_57750130262228
// MI455X (gfx1250) — matrix operand order repaired and checked by reading, not compiled, not run on hardware
//
#include <hip/hip_runtime.h>
#include <hip/hip_bf16.h>
#include <stdint.h>

// ---------------------------------------------------------------------------
// SparseImVoxelNeck for MI455X (gfx1250, wave32, WMMA).
// bf16 activations end-to-end; implicit-GEMM 3x3x3 convs on
// v_wmma_f32_16x16x32_bf16 (12 static WMMAs in the inner stage loop);
// double-buffered ASYNC weight prefetch (global_load_async_to_lds_b128 +
// s_wait_asynccnt) overlapping the WMMA burst; masked-BN statistics fused
// into the conv epilogue (ds_add_f32 + global_atomic_add_f32).
// ---------------------------------------------------------------------------

typedef __attribute__((ext_vector_type(16))) __bf16 v16bf;
typedef __attribute__((ext_vector_type(8)))  float  v8f;
typedef __attribute__((ext_vector_type(4)))  int    v4i;

#if defined(__has_builtin)
#if __has_builtin(__builtin_amdgcn_global_load_async_to_lds_b128) && \
    __has_builtin(__builtin_amdgcn_s_wait_asynccnt)
#define USE_ASYNC 1
#endif
#endif
#ifndef USE_ASYNC
#define USE_ASYNC 0
#endif

// async-to-LDS builtin wants AS(1)/AS(3) pointers to 4xint vectors
typedef __attribute__((address_space(1))) v4i GV4;
typedef __attribute__((address_space(3))) v4i LV4;

#define WDIM   64
#define HWSZ   4096          // 64*64
#define BATCH  2
#define CINCH  64
#define EPSBN  1e-3f

__device__ __forceinline__ unsigned short f2bf(float f) {
  unsigned int u = __float_as_uint(f);
  u += 0x7FFFu + ((u >> 16) & 1u);     // round-to-nearest-even
  return (unsigned short)(u >> 16);
}
__device__ __forceinline__ float bf2f(unsigned short h) {
  return __uint_as_float(((unsigned int)h) << 16);
}

// ---------------------------------------------------------------------------
// Scatter voxel features into dense bf16 grid + fp32 occupancy mask.
// ---------------------------------------------------------------------------
__global__ void k_scatter(const float* __restrict__ vf, const int* __restrict__ coors,
                          unsigned short* __restrict__ X, float* __restrict__ M, int nvox) {
  int t = blockIdx.x * blockDim.x + threadIdx.x;
  if (t >= nvox * CINCH) return;
  int v = t >> 6, c = t & 63;
  int b = coors[v * 4 + 0], z = coors[v * 4 + 1];
  int y = coors[v * 4 + 2], x = coors[v * 4 + 3];
  X[(((size_t)(b * CINCH + c) * 24 + z) * WDIM + y) * WDIM + x] = f2bf(vf[t]);
  if (c == 0) M[(((size_t)b * 24 + z) * WDIM + y) * WDIM + x] = 1.0f;
}

// ---------------------------------------------------------------------------
// Repack OIDHW fp32 weights -> bf16 A-fragments in a contiguous-k per-lane layout
// for v_wmma_*_16x16x32: element (m,k): lane = m + 16*(k>>4),
// slot = k&15.  Buffer: [tap][kc][coTile16][lane][slot].
// ---------------------------------------------------------------------------
__global__ void k_repack(const float* __restrict__ Wt, unsigned short* __restrict__ Ap,
                         int Cout, int total) {
  int i = blockIdx.x * blockDim.x + threadIdx.x;
  if (i >= total) return;
  int tap = i % 27;
  int ci  = (i / 27) & 63;
  int co  = i / (27 * CINCH);
  int k = ci & 31, kc = ci >> 5;
  int hi = k >> 4;
  int sl = k & 15;
  int nCoT16 = Cout >> 4;
  size_t o = ((((size_t)tap * 2 + kc) * nCoT16 + (co >> 4)) * 32 + ((co & 15) + 16 * hi)) * 16 + sl;
  Ap[o] = f2bf(Wt[i]);
}

// ---------------------------------------------------------------------------
// Async (or fallback) prefetch of one stage's 24 A-fragments (24 KB) to LDS.
// Frag f = tapL*8 + kc*4 + coT16local; each frag = 32 lanes x 32 B.
// 1536 16-byte chunks spread over 256 threads (6 each).
// ---------------------------------------------------------------------------
__device__ __forceinline__ void prefetchA(const unsigned short* __restrict__ Ap,
                                          unsigned short* aBuf, int st, int coT64,
                                          int nCoT16, int tid) {
#pragma unroll
  for (int q = 0; q < 6; ++q) {
    const int chunk  = q * 256 + tid;        // 0..1535
    const int f      = chunk >> 6;           // frag 0..23
    const int within = chunk & 63;           // 16B unit within frag
    const int tapL   = f >> 3;
    const int kc     = (f >> 2) & 1;
    const int ct     = f & 3;
    const size_t gfrag = ((size_t)(st * 3 + tapL) * 2 + kc) * nCoT16 + (coT64 * 4 + ct);
    const unsigned short* gp = Ap + gfrag * 512 + within * 8;
    unsigned short* lp = aBuf + f * 512 + within * 8;
#if USE_ASYNC
    __builtin_amdgcn_global_load_async_to_lds_b128((GV4*)gp, (LV4*)lp, 0, 0);
#else
    *(uint4*)lp = *(const uint4*)gp;
#endif
  }
}

// ---------------------------------------------------------------------------
// Implicit-GEMM conv3d (3x3x3, strideH/W=1, padH/W=1, strideD/padD params).
// One workgroup = 64 output channels x 64 output W positions at (b, dOut, h).
// 8 waves, each owns a 16(co) x 32(w) subtile -> 2 f32 accumulators.
// Loop over 9 (kd,kh) stages; each stage stages the bf16 activation row once
// into LDS as THREE shifted B-fragment copies (one per kw) and runs
// 3(kw) x 2(kc) x 2(nt) = 12 WMMAs per wave, while the async unit streams
// the NEXT stage's A-fragments into the other LDS buffer.
// B-fragment layout: lane = n + 16*(k>=16), slot = k&15.
// Epilogue: store raw fp32 conv out + fused masked BN partial sums.
// ---------------------------------------------------------------------------
__global__ __launch_bounds__(256)
void k_conv_wmma(const unsigned short* __restrict__ X, const unsigned short* __restrict__ Ap,
                 float* __restrict__ Y, const float* __restrict__ maskOut,
                 float* __restrict__ s1, float* __restrict__ s2,
                 int Din, int Dout, int Cout, int strideD, int padD) {
  // B: [kw(3)][kc(2)][nt(4)][lane(32)][slot(16)] bf16 = 24 KB
  __shared__ __align__(32) unsigned short ldsB[12288];
  // A: double-buffered stage fragments, 2 x 24 KB
  __shared__ __align__(32) unsigned short aLds[2][12288];

  const int h     = blockIdx.x;
  const int dOut  = blockIdx.y;
  const int nCoT64 = Cout >> 6;
  const int b     = blockIdx.z / nCoT64;
  const int coT64 = blockIdx.z % nCoT64;

  const int tid  = threadIdx.x;
  const int lane = tid & 31;
  const int wv   = tid >> 5;
  const int coSub = wv >> 1;          // 0..3 -> which 16-channel tile
  const int nHalf = wv & 1;           // 0..1 -> which 32-wide half

  // staging role: thread covers one w column x 16 ci
  const int sW  = tid & 63;
  const int sG  = tid >> 6;           // ci block: ci = 16*sG + j  -> k = (sG&1)*16 + j
  const int kcS = sG >> 1;
  const int hiS = sG & 1;
  const int laneS = (sW & 15) + (hiS << 4);
  const int ntS   = sW >> 4;

  const int nCoT16 = Cout >> 4;
  const size_t ciStride = (size_t)Din * HWSZ;

  v8f acc0 = {}; v8f acc1 = {};

  // prologue: bring in stage 0 A-fragments
  prefetchA(Ap, &aLds[0][0], 0, coT64, nCoT16, tid);
#if USE_ASYNC
  __builtin_amdgcn_s_wait_asynccnt(0);
#endif
  __syncthreads();

  for (int st = 0; st < 9; ++st) {
    const int cur = st & 1;
    if (st < 8) prefetchA(Ap, &aLds[cur ^ 1][0], st + 1, coT64, nCoT16, tid);

    const int kd = st / 3, kh = st % 3;
    const int dIn = dOut * strideD + kd - padD;
    const int hIn = h + kh - 1;
    const bool live = (dIn >= 0) & (dIn < Din) & (hIn >= 0) & (hIn < WDIM);  // uniform
    if (live) {
      const unsigned short* col =
          X + (((size_t)(b * CINCH + sG * 16) * Din + dIn) * WDIM + hIn) * WDIM + sW;
      __builtin_prefetch(col + WDIM, 0, 3);   // next kh row -> global_prefetch_b8
      unsigned int pk[3][8] __attribute__((aligned(16)));
#pragma unroll
      for (int j = 0; j < 16; ++j) {
        const unsigned short* r = col + (size_t)j * ciStride;
        unsigned int vm = (sW > 0)        ? (unsigned int)r[-1] : 0u;  // kw=0 (w_in = w-1)
        unsigned int v0 = (unsigned int)r[0];                          // kw=1
        unsigned int vp = (sW < WDIM - 1) ? (unsigned int)r[1] : 0u;   // kw=2
        if (j & 1) {
          pk[0][j >> 1] |= vm << 16; pk[1][j >> 1] |= v0 << 16; pk[2][j >> 1] |= vp << 16;
        } else {
          pk[0][j >> 1] = vm; pk[1][j >> 1] = v0; pk[2][j >> 1] = vp;
        }
      }
#pragma unroll
      for (int c = 0; c < 3; ++c) {
        uint4* dst = (uint4*)&ldsB[(((c * 2 + kcS) * 4 + ntS) * 32 + laneS) * 16];
        dst[0] = *(const uint4*)&pk[c][0];
        dst[1] = *(const uint4*)&pk[c][4];
      }
    }
    __syncthreads();       // B + A[cur] valid; in-flight async touches only A[nxt]
    if (live) {
      const unsigned short* aCur = &aLds[cur][0];
#pragma unroll
      for (int c = 0; c < 3; ++c) {
#pragma unroll
        for (int kc = 0; kc < 2; ++kc) {
          const v16bf a  = *(const v16bf*)(&aCur[(((c * 2 + kc) * 4 + coSub) * 32 + lane) * 16]);
          const v16bf b0 = *(const v16bf*)(&ldsB[(((c * 2 + kc) * 4 + nHalf * 2 + 0) * 32 + lane) * 16]);
          const v16bf b1 = *(const v16bf*)(&ldsB[(((c * 2 + kc) * 4 + nHalf * 2 + 1) * 32 + lane) * 16]);
          acc0 = __builtin_amdgcn_wmma_f32_16x16x32_bf16(false, a, false, b0, (short)0, acc0, false, false);
          acc1 = __builtin_amdgcn_wmma_f32_16x16x32_bf16(false, a, false, b1, (short)0, acc1, false, false);
        }
      }
    }
#if USE_ASYNC
    __builtin_amdgcn_s_wait_asynccnt(0);   // next stage's A landed in LDS
#endif
    __syncthreads();
  }

  // ---- epilogue: store raw fp32 conv output + fused masked-BN partials ----
  float* sLds = (float*)ldsB;                 // reuse LDS: [0..63]=s1, [64..127]=s2
  if (tid < 128) sLds[tid] = 0.0f;
  __syncthreads();

  const float* mrow = maskOut + ((size_t)b * Dout + dOut) * HWSZ + h * WDIM;
  const int mRow8 = (lane >> 4) * 8;          // C/D layout: lanes>=16 hold M+8
  const int nl = lane & 15;
  const int w0 = nHalf * 32 + nl;
  const int w1 = w0 + 16;
  const float m0 = mrow[w0];
  const float m1 = mrow[w1];
#pragma unroll
  for (int r = 0; r < 8; ++r) {
    const int cLoc = coSub * 16 + mRow8 + r;            // 0..63
    const int co   = coT64 * 64 + cLoc;
    const size_t ybase = (((size_t)b * Cout + co) * Dout + dOut) * HWSZ + h * WDIM;
    const float v0 = acc0[r];
    const float v1 = acc1[r];
    Y[ybase + w0] = v0;
    Y[ybase + w1] = v1;
    atomicAdd(&sLds[cLoc],      v0 * m0 + v1 * m1);
    atomicAdd(&sLds[64 + cLoc], v0 * v0 * m0 + v1 * v1 * m1);
  }
  __syncthreads();
  if (tid < 64) {
    atomicAdd(&s1[coT64 * 64 + tid], sLds[tid]);
    atomicAdd(&s2[coT64 * 64 + tid], sLds[64 + tid]);
  }
}

// ---------------------------------------------------------------------------
// Masked BN normalize + scale/shift + ReLU + mask.
// Writes bf16 activation (next conv input); optionally fp32 (final output).
// ---------------------------------------------------------------------------
__global__ void k_bn_apply(const float* __restrict__ Yin, const float* __restrict__ mask,
                           const float* __restrict__ s1, const float* __restrict__ s2,
                           const float* __restrict__ cnt, const float* __restrict__ gamma,
                           const float* __restrict__ beta,
                           unsigned short* __restrict__ outBf, float* __restrict__ outF32,
                           int Cout, int Dout, int total) {
  int i = blockIdx.x * blockDim.x + threadIdx.x;
  if (i >= total) return;
  const int dsz = Dout * HWSZ;
  int sp = i % dsz;
  int c  = (i / dsz) % Cout;
  int b  = i / (dsz * Cout);
  float m    = mask[(size_t)b * dsz + sp];
  float cn   = fmaxf(cnt[0], 1.0f);
  float mean = s1[c] / cn;
  float var  = s2[c] / cn - mean * mean;
  float inv  = rsqrtf(var + EPSBN);
  float y    = (Yin[i] - mean) * inv * gamma[c] + beta[c];
  float res  = (m > 0.0f) ? fmaxf(y, 0.0f) : 0.0f;
  outBf[i] = f2bf(res);
  if (outF32) outF32[i] = res;
}

// x = relu(x + y2) * mask on bf16 activations (C fixed at 64)
__global__ void k_resid(unsigned short* __restrict__ Xio, const unsigned short* __restrict__ Y2,
                        const float* __restrict__ mask, int Dcur, int total) {
  int i = blockIdx.x * blockDim.x + threadIdx.x;
  if (i >= total) return;
  const int dsz = Dcur * HWSZ;
  int sp = i % dsz;
  int b  = i / (dsz * 64);
  float m = mask[(size_t)b * dsz + sp];
  Xio[i] = f2bf(fmaxf(bf2f(Xio[i]) + bf2f(Y2[i]), 0.0f) * m);
}

// mask' = (conv3d(mask, ones 3x3x3, stride (2,1,1), pad (padD,1,1)) > 0)
__global__ void k_maskdown(const float* __restrict__ Min, float* __restrict__ Mout,
                           int Din, int Dout, int padD, int total) {
  int i = blockIdx.x * blockDim.x + threadIdx.x;
  if (i >= total) return;
  int w = i & 63, hh = (i >> 6) & 63;
  int d = (i / HWSZ) % Dout;
  int b = i / (Dout * HWSZ);
  float any = 0.0f;
  for (int kd = 0; kd < 3; ++kd) {
    int di = d * 2 + kd - padD;
    if (di < 0 || di >= Din) continue;
    for (int kh = 0; kh < 3; ++kh) {
      int hi = hh + kh - 1;
      if (hi < 0 || hi >= WDIM) continue;
      for (int kw = 0; kw < 3; ++kw) {
        int wi = w + kw - 1;
        if (wi < 0 || wi >= WDIM) continue;
        if (Min[(((size_t)b * Din + di) * WDIM + hi) * WDIM + wi] > 0.0f) any = 1.0f;
      }
    }
  }
  Mout[i] = any;
}

__global__ void k_masksum(const float* __restrict__ M, float* __restrict__ cnt, int total) {
  __shared__ float red[256];
  int tid = threadIdx.x;
  int i = blockIdx.x * 256 + tid;
  red[tid] = (i < total) ? M[i] : 0.0f;
  __syncthreads();
  for (int s = 128; s > 0; s >>= 1) {
    if (tid < s) red[tid] += red[tid + s];
    __syncthreads();
  }
  if (tid == 0) atomicAdd(cnt, red[0]);
}

// ---------------------------------------------------------------------------
// Host-side orchestration.
// Input flattening assumed jax-pytree order (nested dict keys sorted):
//   0: voxel_features
//   1..3:  conv_input (w,g,b)
//   4..15: down0..down3 (w,g,b each)        [down3: Cout=128]
//   16..39: res0..res3 (wa,ga,ba,wb,gb,bb each)
//   40: coors   41: batch_size
// ---------------------------------------------------------------------------
extern "C" void kernel_launch(void* const* d_in, const int* in_sizes, int n_in,
                              void* d_out, int out_size, void* d_ws, size_t ws_size,
                              hipStream_t stream) {
  (void)n_in; (void)out_size; (void)ws_size;
  const float* vf    = (const float*)d_in[0];
  const int*   coors = (const int*)d_in[40];
  const int    nvox  = in_sizes[0] / CINCH;

  const float* Wci = (const float*)d_in[1];
  const float* Gci = (const float*)d_in[2];
  const float* Bci = (const float*)d_in[3];
  const float *Wdn[4], *Gdn[4], *Bdn[4];
  for (int i = 0; i < 4; ++i) {
    Wdn[i] = (const float*)d_in[4 + 3 * i];
    Gdn[i] = (const float*)d_in[5 + 3 * i];
    Bdn[i] = (const float*)d_in[6 + 3 * i];
  }
  const float *Wra[4], *Gra[4], *Bra[4], *Wrb[4], *Grb[4], *Brb[4];
  for (int i = 0; i < 4; ++i) {
    const int base = 16 + 6 * i;
    Wra[i] = (const float*)d_in[base + 0];
    Gra[i] = (const float*)d_in[base + 1];
    Bra[i] = (const float*)d_in[base + 2];
    Wrb[i] = (const float*)d_in[base + 3];
    Grb[i] = (const float*)d_in[base + 4];
    Brb[i] = (const float*)d_in[base + 5];
  }

  // ---- workspace carve-up ----
  const size_t NELT  = (size_t)BATCH * CINCH * 24 * HWSZ;     // max activation elems
  const size_t BIGH  = NELT * sizeof(unsigned short);         // 25.2 MB (bf16)
  const size_t BIGF  = NELT * sizeof(float);                  // 50.3 MB (fp32 raw conv out)
  const size_t MBYT  = (size_t)BATCH * 24 * HWSZ * sizeof(float);
  size_t off = 0;
  auto carve = [&](size_t bytes) {
    void* p = (char*)d_ws + off;
    off += (bytes + 255) & ~(size_t)255;
    return p;
  };
  unsigned short* p0 = (unsigned short*)carve(BIGH);
  unsigned short* p1 = (unsigned short*)carve(BIGH);
  unsigned short* p2 = (unsigned short*)carve(BIGH);
  float* yraw = (float*)carve(BIGF);
  float* m0 = (float*)carve(MBYT);
  float* m1 = (float*)carve(MBYT);
  unsigned short* apack = (unsigned short*)carve((size_t)128 * CINCH * 27 * 2);
  float* s1p  = (float*)carve(512);
  float* s2p  = (float*)carve(512);
  float* cntp = (float*)carve(256);

  auto blocks = [](int n) { return (n + 255) / 256; };

  // One full conv layer: repack weights, conv (+fused stats), BN apply.
  auto runConv = [&](const unsigned short* Xin, const float* Wraw, const float* maskO,
                     const float* g, const float* bt,
                     unsigned short* bnOutBf, float* bnOutF32,
                     int Din, int Dout, int Cout, int strideD, int padD) {
    const int nW = Cout * CINCH * 27;
    k_repack<<<blocks(nW), 256, 0, stream>>>(Wraw, apack, Cout, nW);
    (void)hipMemsetAsync(s1p, 0, 512, stream);
    (void)hipMemsetAsync(s2p, 0, 512, stream);
    dim3 g3(WDIM, Dout, BATCH * (Cout / 64));
    k_conv_wmma<<<g3, 256, 0, stream>>>(Xin, apack, yraw, maskO, s1p, s2p,
                                        Din, Dout, Cout, strideD, padD);
    const int total = BATCH * Cout * Dout * HWSZ;
    k_bn_apply<<<blocks(total), 256, 0, stream>>>(yraw, maskO, s1p, s2p, cntp,
                                                  g, bt, bnOutBf, bnOutF32,
                                                  Cout, Dout, total);
  };

  // ---- scatter + initial mask count ----
  (void)hipMemsetAsync(p2, 0, BIGH, stream);
  (void)hipMemsetAsync(m0, 0, MBYT, stream);
  k_scatter<<<blocks(nvox * CINCH), 256, 0, stream>>>(vf, coors, p2, m0, nvox);
  (void)hipMemsetAsync(cntp, 0, 4, stream);
  k_masksum<<<blocks(BATCH * 24 * HWSZ), 256, 0, stream>>>(m0, cntp, BATCH * 24 * HWSZ);

  // ---- conv_input ----
  runConv(p2, Wci, m0, Gci, Bci, p0, nullptr, 24, 24, 64, 1, 1);

  unsigned short *x = p0, *t1 = p1, *t2 = p2;
  float *mk = m0, *mkN = m1;
  const int Ds[5] = {24, 12, 6, 3, 1};

  for (int i = 0; i < 4; ++i) {
    const int D = Ds[i], Dn = Ds[i + 1];
    const int padD = (i == 3) ? 0 : 1;
    const int CoD  = (i == 3) ? 128 : 64;

    // residual block
    runConv(x,  Wra[i], mk, Gra[i], Bra[i], t1, nullptr, D, D, 64, 1, 1);
    runConv(t1, Wrb[i], mk, Grb[i], Brb[i], t2, nullptr, D, D, 64, 1, 1);
    const int tot = BATCH * 64 * D * HWSZ;
    k_resid<<<blocks(tot), 256, 0, stream>>>(x, t2, mk, D, tot);

    // mask downsample + new count
    const int mt = BATCH * Dn * HWSZ;
    k_maskdown<<<blocks(mt), 256, 0, stream>>>(mk, mkN, D, Dn, padD, mt);
    (void)hipMemsetAsync(cntp, 0, 4, stream);
    k_masksum<<<blocks(mt), 256, 0, stream>>>(mkN, cntp, mt);

    // strided down conv (final layer also writes fp32 straight to d_out)
    float* f32out = (i == 3) ? (float*)d_out : nullptr;
    runConv(x, Wdn[i], mkN, Gdn[i], Bdn[i], t1, f32out, D, Dn, CoD, 2, padD);

    unsigned short* tf = x; x = t1; t1 = tf;
    float* tm = mk; mk = mkN; mkN = tm;
  }
}
